// QuantizedLinear_9990093931076
// MI455X (gfx1250) — compile-verified, matrix operand order checked by reading, not run on hardware
//
#include <hip/hip_runtime.h>

// CDNA5 (gfx1250) quantized-linear: y = x @ ((Wint - zp)*scale)^T + bias
// Weight-bandwidth-bound (180 MB int32 weights; fits 192 MB L2 on replay).
//
// Identity used:  y = scale*(x @ Wint^T) - scale*zp*rowsum(x) + bias
// so the inner loop converts raw Wint -> fp16 (exact, values in [0,127))
// with NO per-element zero-point subtraction; the zp term is a rank-1
// epilogue correction from per-token row sums.
//
// Pass 1: x fp32 -> fp16 into d_ws + per-token row sums (of the fp16 values).
// Pass 2: single sweep over weights; A-fragments load as f16 (no cvt),
//         one B fragment feeds 4 WMMAs (all 64 tokens); f32 accumulation.

typedef __attribute__((ext_vector_type(16))) _Float16 v16h;
typedef __attribute__((ext_vector_type(8)))  _Float16 h8;
typedef __attribute__((ext_vector_type(4)))  _Float16 h4;
typedef __attribute__((ext_vector_type(8)))  float    v8f;
typedef __attribute__((ext_vector_type(4)))  float    f32x4;
typedef __attribute__((ext_vector_type(4)))  int      i32x4;

#define TOKENS 64   // n_tokens in reference
#define NPB    128  // output columns per block = 8 waves * 16

#define CAT16(lo, hi) __builtin_shufflevector(lo, hi, 0,1,2,3,4,5,6,7,8,9,10,11,12,13,14,15)

// ---------------------------------------------------------------------------
// Pass 1: one block per token row. Convert row to fp16 and reduce its sum
// (of the fp16-rounded values, matching the WMMA accumulation).
// ---------------------------------------------------------------------------
__global__ __launch_bounds__(256)
void x_prep_kernel(const float* __restrict__ x, _Float16* __restrict__ xh,
                   float* __restrict__ xsum, int K)
{
    __shared__ float red[256];
    const int row = blockIdx.x;
    const int tid = threadIdx.x;
    const float*  xr  = x  + (long)row * K;
    _Float16*     xhr = xh + (long)row * K;

    float s = 0.0f;
    for (int k = 4 * tid; k < K; k += 4 * 256) {
        f32x4 v = *(const f32x4*)(xr + k);
        h4 h;
        h[0] = (_Float16)v.x; h[1] = (_Float16)v.y;
        h[2] = (_Float16)v.z; h[3] = (_Float16)v.w;
        *(h4*)(xhr + k) = h;
        s += (float)h[0] + (float)h[1] + (float)h[2] + (float)h[3];
    }
    red[tid] = s;
    __syncthreads();
    for (int off = 128; off > 0; off >>= 1) {
        if (tid < off) red[tid] += red[tid + off];
        __syncthreads();
    }
    if (tid == 0) xsum[row] = red[0];
}

// Raw Wint -> 32x16 fp16 B fragment (exact for [0,127); no zp subtract).
__device__ __forceinline__ v16h make_b(i32x4 w0, i32x4 w1, i32x4 w2, i32x4 w3) {
    v16h b;
    b[0]  = (_Float16)(float)w0.x;  b[1]  = (_Float16)(float)w0.y;
    b[2]  = (_Float16)(float)w0.z;  b[3]  = (_Float16)(float)w0.w;
    b[4]  = (_Float16)(float)w1.x;  b[5]  = (_Float16)(float)w1.y;
    b[6]  = (_Float16)(float)w1.z;  b[7]  = (_Float16)(float)w1.w;
    b[8]  = (_Float16)(float)w2.x;  b[9]  = (_Float16)(float)w2.y;
    b[10] = (_Float16)(float)w2.z;  b[11] = (_Float16)(float)w2.w;
    b[12] = (_Float16)(float)w3.x;  b[13] = (_Float16)(float)w3.y;
    b[14] = (_Float16)(float)w3.z;  b[15] = (_Float16)(float)w3.w;
    return b;
}

// ---------------------------------------------------------------------------
// Pass 2: main GEMM. Each wave owns 16 output columns, all 64 tokens.
// ---------------------------------------------------------------------------
__global__ __launch_bounds__(256)
void qlinear_wmma_f16_kernel(const _Float16* __restrict__ xh,
                             const float*    __restrict__ xsum,
                             const int*      __restrict__ wq,
                             const float*    __restrict__ scale_p,
                             const int*      __restrict__ zp_p,
                             const float*    __restrict__ bias,
                             float*          __restrict__ out,
                             int K, int N)
{
    const int lane = threadIdx.x & 31;
    const int wid  = threadIdx.x >> 5;
    const int col  = lane & 15;
    const int hi   = lane >> 4;

    const int  ncol   = blockIdx.x * NPB + wid * 16 + col;
    const bool valid  = (ncol < N);
    const int  nclamp = valid ? ncol : 0;   // select, not branch: EXEC all-1s for WMMA

    const int   zp    = *zp_p;
    const float scale = *scale_p;

    const int*      wrow  = wq + (long)nclamp * K;
    const _Float16* xrow0 = xh + (long)(col)      * K;
    const _Float16* xrow1 = xh + (long)(16 + col) * K;
    const _Float16* xrow2 = xh + (long)(32 + col) * K;
    const _Float16* xrow3 = xh + (long)(48 + col) * K;

    v8f acc0 = {}, acc1 = {}, acc2 = {}, acc3 = {};

    #pragma unroll 2
    for (int k0 = 0; k0 < K; k0 += 32) {
        const int kb = k0 + 8 * hi;

        // WGP-scope prefetch of the weight stream (locality 3 -> RT, all levels).
        __builtin_prefetch(wrow + kb + 512, 0, 3);

        // ---- issue ALL loads for this K-step up front ----
        i32x4 w0 = *(const i32x4*)(wrow + kb);
        i32x4 w1 = *(const i32x4*)(wrow + kb + 4);
        i32x4 w2 = *(const i32x4*)(wrow + kb + 16);
        i32x4 w3 = *(const i32x4*)(wrow + kb + 20);
        h8 a0lo = *(const h8*)(xrow0 + kb);  h8 a0hi = *(const h8*)(xrow0 + kb + 16);
        h8 a1lo = *(const h8*)(xrow1 + kb);  h8 a1hi = *(const h8*)(xrow1 + kb + 16);
        h8 a2lo = *(const h8*)(xrow2 + kb);  h8 a2hi = *(const h8*)(xrow2 + kb + 16);
        h8 a3lo = *(const h8*)(xrow3 + kb);  h8 a3hi = *(const h8*)(xrow3 + kb + 16);

        // B-convert VALU (24 ops: cvt+pack, no subtracts) under load shadow.
        v16h b = make_b(w0, w1, w2, w3);

        acc0 = __builtin_amdgcn_wmma_f32_16x16x32_f16(false, CAT16(a0lo, a0hi), false, b, (short)0, acc0, false, false);
        acc1 = __builtin_amdgcn_wmma_f32_16x16x32_f16(false, CAT16(a1lo, a1hi), false, b, (short)0, acc1, false, false);
        acc2 = __builtin_amdgcn_wmma_f32_16x16x32_f16(false, CAT16(a2lo, a2hi), false, b, (short)0, acc2, false, false);
        acc3 = __builtin_amdgcn_wmma_f32_16x16x32_f16(false, CAT16(a3lo, a3hi), false, b, (short)0, acc3, false, false);
    }

    if (valid) {
        const float bv = bias[ncol];
        const float zs = scale * (float)zp;   // rank-1 zero-point correction factor
        #pragma unroll
        for (int j = 0; j < 8; ++j) {
            const int t = j + 8 * hi;   // C/D layout: VGPR j -> M = j + 8*(lane>=16)
            out[(long)(t)      * N + ncol] = scale * acc0[j] - zs * xsum[t]      + bv;
            out[(long)(t + 16) * N + ncol] = scale * acc1[j] - zs * xsum[t + 16] + bv;
            out[(long)(t + 32) * N + ncol] = scale * acc2[j] - zs * xsum[t + 32] + bv;
            out[(long)(t + 48) * N + ncol] = scale * acc3[j] - zs * xsum[t + 48] + bv;
        }
    }
}

// ---------------------------------------------------------------------------
// Fallback (workspace too small): fully self-contained, zp folded in-kernel.
// ---------------------------------------------------------------------------
__device__ __forceinline__ v16h load_a_f32(const float* __restrict__ xr, int kb) {
    f32x4 x0 = *(const f32x4*)(xr + kb);
    f32x4 x1 = *(const f32x4*)(xr + kb + 4);
    f32x4 x2 = *(const f32x4*)(xr + kb + 16);
    f32x4 x3 = *(const f32x4*)(xr + kb + 20);
    v16h a;
    a[0]  = (_Float16)x0.x;  a[1]  = (_Float16)x0.y;
    a[2]  = (_Float16)x0.z;  a[3]  = (_Float16)x0.w;
    a[4]  = (_Float16)x1.x;  a[5]  = (_Float16)x1.y;
    a[6]  = (_Float16)x1.z;  a[7]  = (_Float16)x1.w;
    a[8]  = (_Float16)x2.x;  a[9]  = (_Float16)x2.y;
    a[10] = (_Float16)x2.z;  a[11] = (_Float16)x2.w;
    a[12] = (_Float16)x3.x;  a[13] = (_Float16)x3.y;
    a[14] = (_Float16)x3.z;  a[15] = (_Float16)x3.w;
    return a;
}

__device__ __forceinline__ v16h make_b_zp(i32x4 w0, i32x4 w1, i32x4 w2, i32x4 w3, int zp) {
    v16h b;
    b[0]  = (_Float16)(float)(w0.x - zp);  b[1]  = (_Float16)(float)(w0.y - zp);
    b[2]  = (_Float16)(float)(w0.z - zp);  b[3]  = (_Float16)(float)(w0.w - zp);
    b[4]  = (_Float16)(float)(w1.x - zp);  b[5]  = (_Float16)(float)(w1.y - zp);
    b[6]  = (_Float16)(float)(w1.z - zp);  b[7]  = (_Float16)(float)(w1.w - zp);
    b[8]  = (_Float16)(float)(w2.x - zp);  b[9]  = (_Float16)(float)(w2.y - zp);
    b[10] = (_Float16)(float)(w2.z - zp);  b[11] = (_Float16)(float)(w2.w - zp);
    b[12] = (_Float16)(float)(w3.x - zp);  b[13] = (_Float16)(float)(w3.y - zp);
    b[14] = (_Float16)(float)(w3.z - zp);  b[15] = (_Float16)(float)(w3.w - zp);
    return b;
}

__global__ __launch_bounds__(256)
void qlinear_wmma_f32x_kernel(const float* __restrict__ x,
                              const int*   __restrict__ wq,
                              const float* __restrict__ scale_p,
                              const int*   __restrict__ zp_p,
                              const float* __restrict__ bias,
                              float*       __restrict__ out,
                              int K, int N)
{
    const int lane = threadIdx.x & 31;
    const int wid  = threadIdx.x >> 5;
    const int col  = lane & 15;
    const int hi   = lane >> 4;

    const int  ncol   = blockIdx.x * NPB + wid * 16 + col;
    const bool valid  = (ncol < N);
    const int  nclamp = valid ? ncol : 0;

    const int   zp    = *zp_p;
    const float scale = *scale_p;

    const int*   wrow  = wq + (long)nclamp * K;
    const float* xrow0 = x + (long)(col)      * K;
    const float* xrow1 = x + (long)(16 + col) * K;
    const float* xrow2 = x + (long)(32 + col) * K;
    const float* xrow3 = x + (long)(48 + col) * K;

    v8f acc0 = {}, acc1 = {}, acc2 = {}, acc3 = {};

    for (int k0 = 0; k0 < K; k0 += 32) {
        const int kb = k0 + 8 * hi;
        __builtin_prefetch(wrow + kb + 512, 0, 3);

        i32x4 w0 = *(const i32x4*)(wrow + kb);
        i32x4 w1 = *(const i32x4*)(wrow + kb + 4);
        i32x4 w2 = *(const i32x4*)(wrow + kb + 16);
        i32x4 w3 = *(const i32x4*)(wrow + kb + 20);

        v16h b = make_b_zp(w0, w1, w2, w3, zp);

        v16h a0 = load_a_f32(xrow0, kb);
        acc0 = __builtin_amdgcn_wmma_f32_16x16x32_f16(false, a0, false, b, (short)0, acc0, false, false);
        v16h a1 = load_a_f32(xrow1, kb);
        acc1 = __builtin_amdgcn_wmma_f32_16x16x32_f16(false, a1, false, b, (short)0, acc1, false, false);
        v16h a2 = load_a_f32(xrow2, kb);
        acc2 = __builtin_amdgcn_wmma_f32_16x16x32_f16(false, a2, false, b, (short)0, acc2, false, false);
        v16h a3 = load_a_f32(xrow3, kb);
        acc3 = __builtin_amdgcn_wmma_f32_16x16x32_f16(false, a3, false, b, (short)0, acc3, false, false);
    }

    if (valid) {
        const float bv = bias[ncol];
        #pragma unroll
        for (int j = 0; j < 8; ++j) {
            const int t = j + 8 * hi;
            out[(long)(t)      * N + ncol] = scale * acc0[j] + bv;
            out[(long)(t + 16) * N + ncol] = scale * acc1[j] + bv;
            out[(long)(t + 32) * N + ncol] = scale * acc2[j] + bv;
            out[(long)(t + 48) * N + ncol] = scale * acc3[j] + bv;
        }
    }
}

extern "C" void kernel_launch(void* const* d_in, const int* in_sizes, int n_in,
                              void* d_out, int out_size, void* d_ws, size_t ws_size,
                              hipStream_t stream) {
    (void)n_in; (void)out_size;
    const float* x     = (const float*)d_in[0];
    const int*   wq    = (const int*)  d_in[1];
    const float* scale = (const float*)d_in[2];
    const int*   zp    = (const int*)  d_in[3];
    const float* bias  = (const float*)d_in[4];

    const int N = in_sizes[4];            // out_features = 11008
    const int K = in_sizes[0] / TOKENS;   // in_features = 4096
    const int grid = (N + NPB - 1) / NPB; // 86 blocks

    const size_t xh_bytes = (size_t)TOKENS * (size_t)K * sizeof(_Float16); // 512 KB
    const size_t ws_need  = xh_bytes + (size_t)TOKENS * sizeof(float);     // +256 B

    if (ws_size >= ws_need) {
        _Float16* xh   = (_Float16*)d_ws;
        float*    xsum = (float*)((char*)d_ws + xh_bytes);
        x_prep_kernel<<<TOKENS, 256, 0, stream>>>(x, xh, xsum, K);
        qlinear_wmma_f16_kernel<<<grid, 256, 0, stream>>>(
            xh, xsum, wq, scale, zp, bias, (float*)d_out, K, N);
    } else {
        qlinear_wmma_f32x_kernel<<<grid, 256, 0, stream>>>(
            x, wq, scale, zp, bias, (float*)d_out, K, N);
    }
}
